// DeformableCrossAttnBlock_30812095381513
// MI455X (gfx1250) — matrix operand order repaired and checked by reading, not compiled, not run on hardware
//
#include <hip/hip_runtime.h>
#include <hip/hip_bf16.h>

// ---------------------------------------------------------------------------
// DeformableCrossAttnBlock for MI455X (gfx1250, wave32, WMMA + TDM)
//   B=16, C=256, H=W=64, HEADS=8, POINTS=4, D_HEAD=32
// Pipeline:
//   prep_qbf16t : query f32 [b][c][y][x] -> bf16 [b][y][x][c] (LDS transpose)
//   prep_a_conv : w_ow  -> bf16 WMMA-A fragments, K=[tap][ch] (no padding)
//   prep_a_proj : w_proj-> bf16 WMMA-A fragments, 256x256
//   conv_ow_wmma: implicit-GEMM 3x3 conv, one TENSOR_LOAD_TO_LDS per slab row,
//                 B-fragments = single contiguous 32B LDS read,
//                 v_wmma_f32_16x16x32_bf16 (72 K-steps x 6 M-tiles)
//   sample_k    : softmax + bilinear gather + point-weighted sum -> bf16 [pix][c]
//   proj_wmma   : 256x256 GEMM via v_wmma_f32_16x16x32_bf16 + bias
// ---------------------------------------------------------------------------

typedef __attribute__((ext_vector_type(16))) __bf16 v16bf;
typedef __attribute__((ext_vector_type(8)))  float  v8f;
typedef __attribute__((ext_vector_type(8)))  int    v8i;
typedef __attribute__((ext_vector_type(4)))  unsigned int u32x4;
typedef __attribute__((ext_vector_type(8)))  int    i32x8;
typedef __attribute__((ext_vector_type(4)))  int    i32x4;

static __device__ __forceinline__ unsigned short f32_to_bf16(float f) {
    unsigned int u = __float_as_uint(f);
    u = (u + 0x7FFFu + ((u >> 16) & 1u)) >> 16;   // round-to-nearest-even
    return (unsigned short)u;
}

// ---------------------------------------------------------------------------
// Workspace layout (bytes)
// ---------------------------------------------------------------------------
#define WS_A_CONV   0u                 // 6*72*32*16*2    =   442368
#define WS_A_PROJ   442368u            // 16*8*32*16*2    =   131072
#define WS_QB       573440u            // 16*64*64*256*2  = 33554432  [b][y][x][c]
#define WS_OW       34127872u          // 16*96*64*64*4   = 25165824
#define WS_FEAT     59293696u          // 65536*256*2     = 33554432

// ---------------------------------------------------------------------------
// query f32 [b][c][y][x] -> bf16 [b][y][x][c]  (LDS-tiled transpose)
// One block = (b, y, 64-channel chunk): coalesced row reads, coalesced writes.
// ---------------------------------------------------------------------------
__global__ __launch_bounds__(256) void prep_qbf16t(const float* __restrict__ q,
                                                   unsigned short* __restrict__ qb) {
    __shared__ float t[64][65];                   // [c_local][x], +1 pad
    int blk   = blockIdx.x;                       // b*256 + y*4 + chunk
    int chunk = blk & 3;
    int y     = (blk >> 2) & 63;
    int b     = blk >> 8;
    int tid   = threadIdx.x;
#pragma unroll
    for (int i = 0; i < 16; ++i) {
        int e = tid + i * 256;                    // 64*64 elements
        int c = e >> 6, x = e & 63;
        t[c][x] = q[(((size_t)(b * 256 + chunk * 64 + c) * 64 + y) * 64) + x];
    }
    __syncthreads();
#pragma unroll
    for (int i = 0; i < 16; ++i) {
        int e = tid + i * 256;
        int x = e >> 6, c = e & 63;
        qb[(((size_t)(b * 64 + y) * 64 + x) * 256) + chunk * 64 + c] =
            f32_to_bf16(t[c][x]);
    }
}

// ---------------------------------------------------------------------------
// w_ow [96][256][3][3] -> A fragments [mt(6)][sg(72)][lane(32)][j(16)]
// K-step sg = chunk*18 + tap*2 + cs ; k-in-step = j+16*(lane/16)
// channel c = chunk*64 + cs*32 + k-in-step  (one tap per K-step, no padding)
// ---------------------------------------------------------------------------
__global__ __launch_bounds__(512) void prep_a_conv(const float* __restrict__ w_ow,
                                                   unsigned short* __restrict__ A) {
    int blk  = blockIdx.x;                 // mt*72 + sg   (432 blocks)
    int t    = threadIdx.x;
    int lane = t >> 4, j = t & 15;
    int mt   = blk / 72, sg = blk % 72;
    int chunk = sg / 18, r2 = sg % 18;
    int tap  = r2 >> 1, cs = r2 & 1;
    int m    = mt * 16 + (lane & 15);
    int kls  = j + 16 * (lane >> 4);
    int c    = chunk * 64 + cs * 32 + kls;
    A[blk * 512 + t] = f32_to_bf16(w_ow[(m * 256 + c) * 9 + tap]);
}

// w_proj [256][256] -> A fragments [mt(16)][s(8)][lane(32)][j(16)]
__global__ __launch_bounds__(512) void prep_a_proj(const float* __restrict__ w_proj,
                                                   unsigned short* __restrict__ A) {
    int blk  = blockIdx.x;                 // mt*8 + s   (128 blocks)
    int t    = threadIdx.x;
    int lane = t >> 4, j = t & 15;
    int mt   = blk >> 3, s = blk & 7;
    int m    = mt * 16 + (lane & 15);
    int k    = 32 * s + j + 16 * (lane >> 4);
    A[blk * 512 + t] = f32_to_bf16(w_proj[m * 256 + k]);
}

// ---------------------------------------------------------------------------
// Conv 3x3 implicit GEMM.  1 workgroup = one (b,y) row of 64 pixels, 4 waves.
// LDS slab: [row(3)][xslot(66)][c(64)] bf16.  xslot = x+1; slots 0 and 65 are
// permanent-zero halo (never written by TDM); missing edge rows stay zero too.
// B-fragment = 32 contiguous bytes per lane -> one aligned v8i LDS read.
// ---------------------------------------------------------------------------
#define LDS_N (3 * 66 * 64)            // 12672 elements

__global__ __launch_bounds__(128) void conv_ow_wmma(const unsigned short* __restrict__ qb,
                                                    const unsigned short* __restrict__ A,
                                                    const float* __restrict__ b_ow,
                                                    float* __restrict__ ow) {
    __shared__ unsigned short lds_q[LDS_N];

    int b    = blockIdx.x >> 6;
    int y    = blockIdx.x & 63;
    int tid  = threadIdx.x;
    int wave = tid >> 5, lane = tid & 31;
    int n    = lane & 15, half = lane >> 4;
    int xb   = wave * 16 + n;

    v8f acc[6];
#pragma unroll
    for (int i = 0; i < 6; ++i)
#pragma unroll
        for (int r = 0; r < 8; ++r) acc[i][r] = 0.0f;

#if __has_builtin(__builtin_amdgcn_tensor_load_to_lds)
    // One-time zero: halo x-slots + missing edge rows stay 0 forever.
    for (int i = tid; i < LDS_N; i += 128) lds_q[i] = 0;
#endif

    for (int chunk = 0; chunk < 4; ++chunk) {
        __syncthreads();                      // prior-iteration LDS reads done
#if __has_builtin(__builtin_amdgcn_tensor_load_to_lds)
        if (wave == 0) {
            unsigned ldsbase = (unsigned)(uintptr_t)
                (__attribute__((address_space(3))) unsigned short*)lds_q;
            for (int r = 0; r < 3; ++r) {
                int gy = y + r - 1;
                if ((unsigned)gy < 64u) {
                    // 2D tile: 64 contiguous channels (128B) x 64 pixels.
                    unsigned lds_addr = ldsbase + (unsigned)(r * 66 + 1) * 128u;
                    unsigned long long ga = (unsigned long long)(uintptr_t)
                        (qb + ((size_t)(b * 64 + gy) * 64) * 256 + chunk * 64);
                    u32x4 g0;
                    g0.x = 1u;                                  // count=1
                    g0.y = lds_addr;                            // lds_addr
                    g0.z = (unsigned)(ga & 0xFFFFFFFFull);      // global_addr lo
                    g0.w = (unsigned)((ga >> 32) & 0x1FFFFFFull) | (2u << 30);
                    i32x8 g1;
                    g1[0] = (1 << 16);     // data_size = 2 bytes
                    g1[1] = (64 << 16);    // tensor_dim0 = 64
                    g1[2] = (64 << 16);    // tensor_dim1 = 64
                    g1[3] = (64 << 16);    // tile_dim0 = 64 (channels)
                    g1[4] = 64;            // tile_dim1 = 64 (pixels)
                    g1[5] = 256;           // tensor_dim0_stride = pixel pitch
                    g1[6] = 0;
                    g1[7] = 0;
                    i32x4 g2 = {0, 0, 0, 0};
                    i32x4 g3 = {0, 0, 0, 0};
                    i32x8 g4 = {0, 0, 0, 0, 0, 0, 0, 0};
                    __builtin_amdgcn_tensor_load_to_lds(g0, g1, g2, g3, g4, 0);
                }
            }
            __builtin_amdgcn_s_wait_tensorcnt(0);
        }
        __syncthreads();
#else
        // Manual fallback staging, same LDS layout (coalesced in c)
        for (int idx = tid; idx < LDS_N; idx += 128) {
            int row = idx / (66 * 64);
            int rem = idx - row * (66 * 64);
            int xs = rem >> 6, c = rem & 63;
            int gy = y + row - 1, x = xs - 1;
            unsigned short v = 0;
            if ((unsigned)gy < 64u && (unsigned)x < 64u)
                v = qb[(((size_t)(b * 64 + gy) * 64 + x) * 256) + chunk * 64 + c];
            lds_q[idx] = v;
        }
        __syncthreads();
#endif

#pragma unroll
        for (int tap = 0; tap < 9; ++tap) {
            const int ty = tap / 3, tx = tap % 3;
#pragma unroll
            for (int cs = 0; cs < 2; ++cs) {
                int sg = chunk * 18 + tap * 2 + cs;
                // preload all 6 A fragments (clause-friendly, one wait)
                v8i araw[6];
#pragma unroll
                for (int mt = 0; mt < 6; ++mt)
                    araw[mt] = *(const v8i*)(A + ((size_t)(mt * 72 + sg) * 32 + lane) * 16);
                // B fragment: 16 contiguous channels at (row=ty, xslot=xb+tx)
                int be = ((ty * 66 + xb + tx) * 64) + cs * 32 + 16 * half;
                v8i braw = *(const v8i*)(lds_q + be);
                v16bf Bf = __builtin_bit_cast(v16bf, braw);
#pragma unroll
                for (int mt = 0; mt < 6; ++mt) {
                    v16bf Af = __builtin_bit_cast(v16bf, araw[mt]);
                    acc[mt] = __builtin_amdgcn_wmma_f32_16x16x32_bf16(
                        false, Af, false, Bf, (short)0, acc[mt], false, false);
                }
            }
        }
    }

    // D[m][n] -> ow[b][m][y][x],  m = 16*mt + r + 8*half, x = xb
#pragma unroll
    for (int mt = 0; mt < 6; ++mt)
#pragma unroll
        for (int r = 0; r < 8; ++r) {
            int m = mt * 16 + r + 8 * half;
            ow[(((size_t)b * 96 + m) * 64 + y) * 64 + xb] = acc[mt][r] + b_ow[m];
        }
}

// ---------------------------------------------------------------------------
// Softmax over points + bilinear sampling + weighted channel sum.
// One thread = one (b, head, y, x); writes feat transposed [pix][c] in bf16.
// ---------------------------------------------------------------------------
__global__ __launch_bounds__(256) void sample_k(const float* __restrict__ ow,
                                                const float* __restrict__ key,
                                                unsigned short* __restrict__ feat_t) {
    int gid = blockIdx.x * 256 + threadIdx.x;
    int x = gid & 63, y = (gid >> 6) & 63, h = (gid >> 12) & 7, b = gid >> 15;

    const float* owp = ow + (size_t)b * 96 * 4096 + y * 64 + x;
    float offx[4], offy[4], wl[4];
#pragma unroll
    for (int p = 0; p < 4; ++p) {
        int ch  = (h * 4 + p) * 3;
        offx[p] = owp[(size_t)(ch + 0) * 4096];
        offy[p] = owp[(size_t)(ch + 1) * 4096];
        wl[p]   = owp[(size_t)(ch + 2) * 4096];
    }
    float mx = fmaxf(fmaxf(wl[0], wl[1]), fmaxf(wl[2], wl[3]));
    float sum = 0.0f;
#pragma unroll
    for (int p = 0; p < 4; ++p) { wl[p] = __expf(wl[p] - mx); sum += wl[p]; }
    float inv = 1.0f / sum;

    float acc[32];
#pragma unroll
    for (int d = 0; d < 32; ++d) acc[d] = 0.0f;

    const float* keyp = key + (size_t)(b * 256 + h * 32) * 4096;
#pragma unroll 1
    for (int p = 0; p < 4; ++p) {
        float wp = wl[p] * inv;
        float px = (float)x + 3.15f * offx[p];   // (loc+1)*0.5*(W-1) simplified
        float py = (float)y + 3.15f * offy[p];
        float x0f = floorf(px), y0f = floorf(py);
        int   x0 = (int)x0f, y0 = (int)y0f;
        float wx = px - x0f, wy = py - y0f;
#pragma unroll
        for (int cy = 0; cy < 2; ++cy)
#pragma unroll
            for (int cx = 0; cx < 2; ++cx) {
                int   xi = x0 + cx, yi = y0 + cy;
                float cw = (cx ? wx : 1.0f - wx) * (cy ? wy : 1.0f - wy) * wp;
                if ((unsigned)xi < 64u && (unsigned)yi < 64u) {
                    const float* kp = keyp + yi * 64 + xi;
#pragma unroll
                    for (int d = 0; d < 32; ++d) acc[d] += cw * kp[(size_t)d * 4096];
                }
            }
    }

    unsigned short* fp = feat_t + ((size_t)(b * 4096 + y * 64 + x)) * 256 + h * 32;
#pragma unroll
    for (int d = 0; d < 32; ++d) fp[d] = f32_to_bf16(acc[d]);
}

// ---------------------------------------------------------------------------
// Projection GEMM: out[o][pix] = sum_c w_proj[o][c]*feat[c][pix] + b_proj[o]
// Wave owns 16 pixels; all 8 B K-step fragments live in registers;
// per M-tile: preload 8 A fragments, then 8 back-to-back WMMAs.
// ---------------------------------------------------------------------------
__global__ __launch_bounds__(128) void proj_wmma(const unsigned short* __restrict__ feat_t,
                                                 const unsigned short* __restrict__ A,
                                                 const float* __restrict__ b_proj,
                                                 float* __restrict__ out) {
    int tid  = threadIdx.x;
    int wave = tid >> 5, lane = tid & 31;
    int n    = lane & 15, half = lane >> 4;
    int pix0 = blockIdx.x * 64 + wave * 16;

    const unsigned short* base = feat_t + (size_t)(pix0 + n) * 256 + half * 16;
    v16bf Bf[8];
#pragma unroll
    for (int s = 0; s < 8; ++s) {
        v8i raw = *(const v8i*)(base + s * 32);
        Bf[s] = __builtin_bit_cast(v16bf, raw);
    }

    int pix = pix0 + n;
    int b   = pix >> 12, yx = pix & 4095;

#pragma unroll 1
    for (int mt = 0; mt < 16; ++mt) {
        if (mt + 1 < 16)
            __builtin_prefetch(A + ((size_t)((mt + 1) * 8) * 32 + lane) * 16, 0, 1);
        v8i araw[8];
#pragma unroll
        for (int s = 0; s < 8; ++s)
            araw[s] = *(const v8i*)(A + ((size_t)(mt * 8 + s) * 32 + lane) * 16);
        v8f acc;
#pragma unroll
        for (int r = 0; r < 8; ++r) acc[r] = 0.0f;
#pragma unroll
        for (int s = 0; s < 8; ++s) {
            v16bf Af = __builtin_bit_cast(v16bf, araw[s]);
            acc = __builtin_amdgcn_wmma_f32_16x16x32_bf16(
                false, Af, false, Bf[s], (short)0, acc, false, false);
        }
#pragma unroll
        for (int r = 0; r < 8; ++r) {
            int m = mt * 16 + r + 8 * half;
            out[((size_t)(b * 256 + m)) * 4096 + yx] = acc[r] + b_proj[m];
        }
    }
}

// ---------------------------------------------------------------------------
extern "C" void kernel_launch(void* const* d_in, const int* in_sizes, int n_in,
                              void* d_out, int out_size, void* d_ws, size_t ws_size,
                              hipStream_t stream) {
    (void)in_sizes; (void)n_in; (void)out_size; (void)ws_size;

    const float* query  = (const float*)d_in[0];
    const float* key    = (const float*)d_in[1];
    const float* w_ow   = (const float*)d_in[2];
    const float* b_ow   = (const float*)d_in[3];
    const float* w_proj = (const float*)d_in[4];
    const float* b_proj = (const float*)d_in[5];
    float*       out    = (float*)d_out;

    char* ws = (char*)d_ws;
    unsigned short* A_conv = (unsigned short*)(ws + WS_A_CONV);
    unsigned short* A_proj = (unsigned short*)(ws + WS_A_PROJ);
    unsigned short* qb     = (unsigned short*)(ws + WS_QB);
    float*          ow     = (float*)(ws + WS_OW);
    unsigned short* feat_t = (unsigned short*)(ws + WS_FEAT);

    prep_qbf16t<<<16 * 64 * 4, 256, 0, stream>>>(query, qb);
    prep_a_conv<<<432, 512, 0, stream>>>(w_ow, A_conv);
    prep_a_proj<<<128, 512, 0, stream>>>(w_proj, A_proj);
    conv_ow_wmma<<<16 * 64, 128, 0, stream>>>(qb, A_conv, b_ow, ow);
    sample_k<<<(16 * 8 * 64 * 64) / 256, 256, 0, stream>>>(ow, key, feat_t);
    proj_wmma<<<(16 * 64 * 64) / 64, 128, 0, stream>>>(feat_t, A_proj, b_proj, out);
}
